// Model_47047071761046
// MI455X (gfx1250) — compile-verified, matrix operand order checked by reading, not run on hardware
//
#include <hip/hip_runtime.h>
#include <math.h>

typedef __attribute__((ext_vector_type(16))) _Float16 v16h;
typedef __attribute__((ext_vector_type(8)))  _Float16 v8h;
typedef __attribute__((ext_vector_type(8)))  float    v8f;

#define NNODES 50000
#define NEDGES 1600000
#define ETOT   (NEDGES + NNODES)
#define BN_EPS 1e-5f
#define NEG_SLOPE 0.2f

// ---- order-preserving float <-> uint key for atomicMax on floats ----
__device__ __forceinline__ unsigned fkey(float f) {
  unsigned b = __float_as_uint(f);
  return (b & 0x80000000u) ? ~b : (b | 0x80000000u);
}
__device__ __forceinline__ float funkey(unsigned k) {
  unsigned b = (k & 0x80000000u) ? (k ^ 0x80000000u) : ~k;
  return __uint_as_float(b);
}

// ---------------- f32 -> f16 convert (activations) ----------------
__global__ void k_f32_to_f16(const float* __restrict__ in, _Float16* __restrict__ out, int n) {
  int i = blockIdx.x * blockDim.x + threadIdx.x;
  if (i < n) out[i] = (_Float16)in[i];
}

// ---------------- pack W[K,Nc] (f32, row-major) into WMMA B-fragment order ----
// Packed index: (((tn*kSteps + ks)*32 + lane)*16 + i)  -> 16 contiguous halves/lane
__global__ void k_pack_w(const float* __restrict__ W, _Float16* __restrict__ Bp, int K, int Nc) {
  int t = blockIdx.x * blockDim.x + threadIdx.x;
  if (t >= K * Nc) return;
  int i    = t & 15;
  int lane = (t >> 4) & 31;
  int rest = t >> 9;              // tn*kSteps + ks
  int kSteps = K >> 5;
  int ks = rest % kSteps;
  int tn = rest / kSteps;
  int laneLo = lane & 15, laneHi = lane >> 4;
  // 16-bit B 32x16 layout: elems 0..7 -> K=0..7, 8..15 -> K=16..23; lanes 16-31: +8
  int k   = ks * 32 + ((i & 8) ? (i + 8) : i) + (laneHi ? 8 : 0);
  int col = tn * 16 + laneLo;
  Bp[t] = (_Float16)W[k * Nc + col];
}

// ---------------- WMMA GEMM: C[M,Nc] = A[M,K] * Bpacked ----------------
// One wave per 16-row strip; TN = Nc/16 column tiles share one A fragment.
template <int TN>
__global__ void k_gemm_wmma(const _Float16* __restrict__ A, const _Float16* __restrict__ Bp,
                            float* __restrict__ C, int K, int Nc, int numRowTiles) {
  int tm = blockIdx.x * (blockDim.x >> 5) + (threadIdx.x >> 5);
  if (tm >= numRowTiles) return;            // whole-wave uniform exit -> EXEC all-1s inside
  int lane   = threadIdx.x & 31;
  int laneLo = lane & 15, laneHi = lane >> 4;
  int row    = tm * 16 + laneLo;
  int kSteps = K >> 5;

  v8f zero = {};
  v8f acc[TN];
#pragma unroll
  for (int t = 0; t < TN; ++t) acc[t] = zero;

  for (int ks = 0; ks < kSteps; ++ks) {
    const _Float16* ap = A + (size_t)row * K + ks * 32 + laneHi * 8;
    v8h alo = *(const v8h*)(ap);
    v8h ahi = *(const v8h*)(ap + 16);
    v16h a;
#pragma unroll
    for (int i = 0; i < 8; ++i) { a[i] = alo[i]; a[i + 8] = ahi[i]; }
#pragma unroll
    for (int t = 0; t < TN; ++t) {
      v16h b = *(const v16h*)(Bp + ((size_t)(t * kSteps + ks) * 32 + lane) * 16);
      acc[t] = __builtin_amdgcn_wmma_f32_16x16x32_f16(false, a, false, b, (short)0,
                                                      acc[t], false, false);
    }
  }
#pragma unroll
  for (int t = 0; t < TN; ++t)
#pragma unroll
    for (int r = 0; r < 8; ++r)
      C[(size_t)(tm * 16 + r + laneHi * 8) * Nc + t * 16 + laneLo] = acc[t][r];
}

// ---------------- per-(node,head) attention logits ----------------
__global__ void k_node_logits(const float* __restrict__ h, const float* __restrict__ a_s,
                              const float* __restrict__ a_d, float* __restrict__ als,
                              float* __restrict__ ald, int n, int H, int Cc) {
  int t = blockIdx.x * blockDim.x + threadIdx.x;
  if (t >= n * H) return;
  int node = t / H, hd = t % H;
  const float* hp = h + (size_t)node * H * Cc + hd * Cc;
  float s = 0.f, d0 = 0.f;
  for (int c = 0; c < Cc; ++c) { float v = hp[c]; s += v * a_s[hd * Cc + c]; d0 += v * a_d[hd * Cc + c]; }
  als[t] = s; ald[t] = d0;
}

// ---------------- init buffers ----------------
__global__ void k_init_stats(unsigned* __restrict__ mkey, float* __restrict__ denom, int n) {
  int i = blockIdx.x * blockDim.x + threadIdx.x;
  if (i < n) { mkey[i] = 0x007FFFFFu; /* fkey(-inf) */ denom[i] = 0.f; }
}
__global__ void k_zero(float* __restrict__ p, int n) {
  int i = blockIdx.x * blockDim.x + threadIdx.x;
  if (i < n) p[i] = 0.f;
}

// ---------------- edge pass 1: segment max ----------------
__global__ void k_edge_max(const int* __restrict__ ei, const float* __restrict__ als,
                           const float* __restrict__ ald, unsigned* __restrict__ mkey, int H) {
  int e = blockIdx.x * blockDim.x + threadIdx.x;
  if (e >= ETOT) return;
  int s, d;
  if (e < NEDGES) { s = ei[e]; d = ei[NEDGES + e]; } else { s = d = e - NEDGES; }
  for (int hd = 0; hd < H; ++hd) {
    float v = als[s * H + hd] + ald[d * H + hd];
    v = (v >= 0.f) ? v : NEG_SLOPE * v;
    atomicMax(&mkey[d * H + hd], fkey(v));
  }
}

// ---------------- edge pass 2: segment sum of exp ----------------
__global__ void k_edge_denom(const int* __restrict__ ei, const float* __restrict__ als,
                             const float* __restrict__ ald, const unsigned* __restrict__ mkey,
                             float* __restrict__ denom, int H) {
  int e = blockIdx.x * blockDim.x + threadIdx.x;
  if (e >= ETOT) return;
  int s, d;
  if (e < NEDGES) { s = ei[e]; d = ei[NEDGES + e]; } else { s = d = e - NEDGES; }
  for (int hd = 0; hd < H; ++hd) {
    float v = als[s * H + hd] + ald[d * H + hd];
    v = (v >= 0.f) ? v : NEG_SLOPE * v;
    atomicAdd(&denom[d * H + hd], __expf(v - funkey(mkey[d * H + hd])));
  }
}

// ---------------- edge pass 3: weighted scatter-add, one wave per edge ----------------
__global__ void k_edge_agg(const int* __restrict__ ei, const float* __restrict__ als,
                           const float* __restrict__ ald, const unsigned* __restrict__ mkey,
                           const float* __restrict__ denom, const float* __restrict__ h,
                           float* __restrict__ agg, int H, int HC) {
  int e = blockIdx.x * (blockDim.x >> 5) + (threadIdx.x >> 5);
  int lane = threadIdx.x & 31;
  if (e >= ETOT) return;
  int s, d;
  if (e < NEDGES) { s = ei[e]; d = ei[NEDGES + e]; } else { s = d = e - NEDGES; }
  for (int c = lane; c < HC; c += 32) {
    int hd = c >> 4;
    float v = als[s * H + hd] + ald[d * H + hd];
    v = (v >= 0.f) ? v : NEG_SLOPE * v;
    float alpha = __expf(v - funkey(mkey[d * H + hd])) / (denom[d * H + hd] + 1e-16f);
    atomicAdd(&agg[(size_t)d * HC + c], alpha * h[(size_t)s * HC + c]);
  }
}

// ---------------- bias + batchnorm + relu ----------------
__global__ void k_bn_relu(const float* __restrict__ agg, const float* __restrict__ bias,
                          const float* __restrict__ g, const float* __restrict__ bb,
                          const float* __restrict__ mu, const float* __restrict__ var,
                          float* __restrict__ out, int total, int HC) {
  int i = blockIdx.x * blockDim.x + threadIdx.x;
  if (i >= total) return;
  int c = i % HC;
  float v = agg[i] + bias[c];
  v = (v - mu[c]) * rsqrtf(var[c] + BN_EPS) * g[c] + bb[c];
  out[i] = (v > 0.f) ? v : 0.f;
}

// ---------------- final MLP head: 16 -> 8 (relu) -> 1 ----------------
__global__ void k_mlp(const float* __restrict__ feat, const float* __restrict__ w1,
                      const float* __restrict__ b1, const float* __restrict__ w2,
                      const float* __restrict__ b2, float* __restrict__ out) {
  int n = blockIdx.x * blockDim.x + threadIdx.x;
  if (n >= NNODES) return;
  float x[16];
#pragma unroll
  for (int i = 0; i < 16; ++i) x[i] = feat[(size_t)n * 16 + i];
  float acc = b2[0];
#pragma unroll
  for (int j = 0; j < 8; ++j) {
    float s = b1[j];
#pragma unroll
    for (int i = 0; i < 16; ++i) s += x[i] * w1[i * 8 + j];
    s = (s > 0.f) ? s : 0.f;
    acc += s * w2[j];
  }
  out[n] = acc;
}

extern "C" void kernel_launch(void* const* d_in, const int* in_sizes, int n_in,
                              void* d_out, int out_size, void* d_ws, size_t ws_size,
                              hipStream_t stream) {
  const float* x  = (const float*)d_in[0];
  const int*   ei = (const int*)d_in[1];
  const float* W[3]  = {(const float*)d_in[2],  (const float*)d_in[10], (const float*)d_in[18]};
  const float* As[3] = {(const float*)d_in[3],  (const float*)d_in[11], (const float*)d_in[19]};
  const float* Ad[3] = {(const float*)d_in[4],  (const float*)d_in[12], (const float*)d_in[20]};
  const float* Bi[3] = {(const float*)d_in[5],  (const float*)d_in[13], (const float*)d_in[21]};
  const float* Bg[3] = {(const float*)d_in[6],  (const float*)d_in[14], (const float*)d_in[22]};
  const float* Bb[3] = {(const float*)d_in[7],  (const float*)d_in[15], (const float*)d_in[23]};
  const float* Bm[3] = {(const float*)d_in[8],  (const float*)d_in[16], (const float*)d_in[24]};
  const float* Bv[3] = {(const float*)d_in[9],  (const float*)d_in[17], (const float*)d_in[25]};
  const float* fc1w = (const float*)d_in[26]; const float* fc1b = (const float*)d_in[27];
  const float* fc2w = (const float*)d_in[28]; const float* fc2b = (const float*)d_in[29];

  // carve workspace
  char* p = (char*)d_ws;
  auto take = [&](size_t bytes) -> char* {
    char* r = p; p += (bytes + 255) & ~(size_t)255; return r;
  };
  _Float16* xh  = (_Float16*)take((size_t)NNODES * 128 * 2);
  _Float16* wp  = (_Float16*)take((size_t)128 * 64 * 2);
  float*    h   = (float*)take((size_t)NNODES * 64 * 4);
  float*    agg = (float*)take((size_t)NNODES * 64 * 4);
  float*    ft  = (float*)take((size_t)NNODES * 64 * 4);
  float*    als = (float*)take((size_t)NNODES * 4 * 4);
  float*    ald = (float*)take((size_t)NNODES * 4 * 4);
  unsigned* mk  = (unsigned*)take((size_t)NNODES * 4 * 4);
  float*    den = (float*)take((size_t)NNODES * 4 * 4);

  const int indim[3] = {128, 64, 64};
  const int Hh[3]    = {4, 4, 1};
  const int Cc = 16;
  const int numRowTiles = NNODES / 16;   // 3125 exactly

  const float* inf = x;
  for (int L = 0; L < 3; ++L) {
    int K = indim[L], H = Hh[L], HC = H * Cc;
    int nx = NNODES * K;
    k_f32_to_f16<<<(nx + 255) / 256, 256, 0, stream>>>(inf, xh, nx);
    int nw = K * HC;
    k_pack_w<<<(nw + 255) / 256, 256, 0, stream>>>(W[L], wp, K, HC);

    if (HC == 64)
      k_gemm_wmma<4><<<(numRowTiles + 3) / 4, 128, 0, stream>>>(xh, wp, h, K, HC, numRowTiles);
    else
      k_gemm_wmma<1><<<(numRowTiles + 3) / 4, 128, 0, stream>>>(xh, wp, h, K, HC, numRowTiles);

    k_node_logits<<<(NNODES * H + 255) / 256, 256, 0, stream>>>(h, As[L], Ad[L], als, ald, NNODES, H, Cc);
    k_init_stats<<<(NNODES * H + 255) / 256, 256, 0, stream>>>(mk, den, NNODES * H);
    k_zero<<<(NNODES * HC + 255) / 256, 256, 0, stream>>>(agg, NNODES * HC);

    k_edge_max  <<<(ETOT + 255) / 256, 256, 0, stream>>>(ei, als, ald, mk, H);
    k_edge_denom<<<(ETOT + 255) / 256, 256, 0, stream>>>(ei, als, ald, mk, den, H);
    k_edge_agg  <<<(ETOT + 7) / 8,     256, 0, stream>>>(ei, als, ald, mk, den, h, agg, H, HC);

    k_bn_relu<<<(NNODES * HC + 255) / 256, 256, 0, stream>>>(agg, Bi[L], Bg[L], Bb[L], Bm[L], Bv[L],
                                                             ft, NNODES * HC, HC);
    inf = ft;
  }
  k_mlp<<<(NNODES + 255) / 256, 256, 0, stream>>>(ft, fc1w, fc1b, fc2w, fc2b, (float*)d_out);
}
